// ClassifierExtWithBefore_2010044694969
// MI455X (gfx1250) — matrix operand order repaired and checked by reading, not compiled, not run on hardware
//
#include <hip/hip_runtime.h>
#include <cstdint>
#include <cstddef>

// ---------------------------------------------------------------------------
// ClassifierExtWithBefore on MI455X (gfx1250, wave32, WMMA f32<-f16 16x16x32)
//
// sent = x @ W1 + b1            GEMM1: M=32768 N=1024 K=3072  (wmma f16)
// before = exclusive cummax_S    elementwise scan, fp32
// h = relu([sent,before]@W2+b2)  GEMM2: M=32768 N=1024 K=2048  (wmma f16)
// logits = h @ W3 + b3           fused into GEMM2 epilogue (shfl + atomicAdd)
// out = softmax_K(logits)*mask   tiny kernel
// ---------------------------------------------------------------------------

typedef __attribute__((ext_vector_type(16))) _Float16 v16h;
typedef __attribute__((ext_vector_type(8)))  _Float16 v8h;
typedef __attribute__((ext_vector_type(8)))  float    v8f;

#define HDIM  1024
#define BB    8
#define SS    512
#define KC    8                    // classes per token
#define MROWS (BB * SS * KC)       // 32768
#define KD1   (3 * HDIM)           // 3072
#define KD2   (2 * HDIM)           // 2048

// ---------------- fragment loaders (wave32 WMMA layouts, ISA 7.12.2) -------

// A 16x32 f16: lane L -> row M = L%16, element i -> K = (i<8 ? i : i+8) + (L/16)*8
__device__ __forceinline__ v16h load_a_f32(const float* __restrict__ base,
                                           int ld, int row, int k0, int lane) {
  const int hf = lane >> 4;
  const int m  = lane & 15;
  const float* p = base + (size_t)(row + m) * ld + k0 + hf * 8;
  v16h f;
#pragma unroll
  for (int i = 0; i < 8; ++i) f[i]     = (_Float16)p[i];
#pragma unroll
  for (int i = 0; i < 8; ++i) f[8 + i] = (_Float16)p[16 + i];
  return f;
}

__device__ __forceinline__ v16h load_a_h(const _Float16* __restrict__ base,
                                         int ld, int row, int k0, int lane) {
  const int hf = lane >> 4;
  const int m  = lane & 15;
  const _Float16* p = base + (size_t)(row + m) * ld + k0 + hf * 8;
  v8h lo = *(const v8h*)p;
  v8h hi = *(const v8h*)(p + 16);
  v16h f;
#pragma unroll
  for (int i = 0; i < 8; ++i) { f[i] = lo[i]; f[8 + i] = hi[i]; }
  return f;
}

// B 32x16 f16: lane L -> column N = L%16, element i -> K = (i<8 ? i : i+8) + (L/16)*8.
// bt is the pre-transposed weight, N x K row-major -> two 16-byte runs/lane.
__device__ __forceinline__ v16h load_b_h(const _Float16* __restrict__ bt,
                                         int ld, int col, int k0, int lane) {
  const int hf = lane >> 4;
  const int n  = lane & 15;
  const _Float16* p = bt + (size_t)(col + n) * ld + k0 + hf * 8;
  v8h lo = *(const v8h*)p;
  v8h hi = *(const v8h*)(p + 16);
  v16h f;
#pragma unroll
  for (int i = 0; i < 8; ++i) { f[i] = lo[i]; f[8 + i] = hi[i]; }
  return f;
}

// ---------------- weight transpose + fp32->f16 convert ---------------------
// w: rows x cols fp32 (row = K dim, col = N dim); wt: cols x rows f16
__global__ __launch_bounds__(256)
void transpose_to_half_kernel(const float* __restrict__ w,
                              _Float16* __restrict__ wt,
                              int rows, int cols) {
  __shared__ float tile[32][33];
  const int c0 = blockIdx.x * 32;
  const int r0 = blockIdx.y * 32;
  const int tx = threadIdx.x;      // 0..31
  const int ty = threadIdx.y;      // 0..7
#pragma unroll
  for (int i = 0; i < 32; i += 8)
    tile[ty + i][tx] = w[(size_t)(r0 + ty + i) * cols + (c0 + tx)];
  __syncthreads();
#pragma unroll
  for (int i = 0; i < 32; i += 8)
    wt[(size_t)(c0 + ty + i) * rows + (r0 + tx)] = (_Float16)tile[tx][ty + i];
}

// ---------------- GEMM1: sent = f16(x) @ f16(W1) + b1  (fp32 out) ----------
// block tile 128(M) x 128(N), 8 waves as 4(M) x 2(N), wave tile 32x64.
__global__ __launch_bounds__(256)
void gemm1_kernel(const float* __restrict__ x, const _Float16* __restrict__ w1t,
                  const float* __restrict__ b1, float* __restrict__ sent) {
  const int lane  = threadIdx.x & 31;
  const int wave  = threadIdx.x >> 5;
  const int wm    = wave >> 1;                  // 0..3
  const int wn    = wave & 1;                   // 0..1
  const int mBase = blockIdx.y * 128 + wm * 32;
  const int nBase = blockIdx.x * 128 + wn * 64;

  const v8f zero = {0.f, 0.f, 0.f, 0.f, 0.f, 0.f, 0.f, 0.f};
  v8f acc[2][4];
#pragma unroll
  for (int i = 0; i < 2; ++i)
#pragma unroll
    for (int j = 0; j < 4; ++j) acc[i][j] = zero;

  for (int k0 = 0; k0 < KD1; k0 += 32) {
    v16h a0 = load_a_f32(x, KD1, mBase,      k0, lane);
    v16h a1 = load_a_f32(x, KD1, mBase + 16, k0, lane);
#pragma unroll
    for (int j = 0; j < 4; ++j) {
      v16h b = load_b_h(w1t, KD1, nBase + j * 16, k0, lane);
      acc[0][j] = __builtin_amdgcn_wmma_f32_16x16x32_f16(
          false, a0, false, b, (short)0, acc[0][j], false, false);
      acc[1][j] = __builtin_amdgcn_wmma_f32_16x16x32_f16(
          false, a1, false, b, (short)0, acc[1][j], false, false);
    }
  }

  // C/D layout: lane 0-15 -> N=lane, VGPR r -> M = r + 8*(lane/16)
  const int hf = lane >> 4;
  const int nl = lane & 15;
#pragma unroll
  for (int j = 0; j < 4; ++j) {
    const int col   = nBase + j * 16 + nl;
    const float bia = b1[col];
#pragma unroll
    for (int mi = 0; mi < 2; ++mi)
#pragma unroll
      for (int r = 0; r < 8; ++r) {
        const int row = mBase + mi * 16 + hf * 8 + r;
        sent[(size_t)row * HDIM + col] = acc[mi][j][r] + bia;
      }
  }
}

// ---------------- exclusive cummax along S, emit cur = [f16 sent | f16 before]
__global__ __launch_bounds__(256)
void cummax_kernel(const float* __restrict__ sent, _Float16* __restrict__ cur) {
  const int t = blockIdx.x * blockDim.x + threadIdx.x;   // over BB*KC*HDIM
  if (t >= BB * KC * HDIM) return;
  const int h = t % HDIM;
  const int k = (t / HDIM) % KC;
  const int b = t / (HDIM * KC);
  float run = 0.f;
  for (int s = 0; s < SS; ++s) {
    const size_t row = (size_t)(b * SS + s) * KC + k;
    const float v      = sent[row * HDIM + h];
    const float before = (s == 0) ? 0.f : run;
    run = (s == 0) ? v : fmaxf(run, v);
    cur[row * KD2 + h]        = (_Float16)v;
    cur[row * KD2 + HDIM + h] = (_Float16)before;
  }
}

// ---------------- GEMM2: relu(cur@W2+b2) . W3  -> atomic logits ------------
__global__ __launch_bounds__(256)
void gemm2_kernel(const _Float16* __restrict__ cur,
                  const _Float16* __restrict__ w2t,
                  const float* __restrict__ b2, const float* __restrict__ w3,
                  float* __restrict__ logits) {
  const int lane  = threadIdx.x & 31;
  const int wave  = threadIdx.x >> 5;
  const int wm    = wave >> 1;
  const int wn    = wave & 1;
  const int mBase = blockIdx.y * 128 + wm * 32;
  const int nBase = blockIdx.x * 128 + wn * 64;

  const v8f zero = {0.f, 0.f, 0.f, 0.f, 0.f, 0.f, 0.f, 0.f};
  v8f acc[2][4];
#pragma unroll
  for (int i = 0; i < 2; ++i)
#pragma unroll
    for (int j = 0; j < 4; ++j) acc[i][j] = zero;

  for (int k0 = 0; k0 < KD2; k0 += 32) {
    v16h a0 = load_a_h(cur, KD2, mBase,      k0, lane);
    v16h a1 = load_a_h(cur, KD2, mBase + 16, k0, lane);
#pragma unroll
    for (int j = 0; j < 4; ++j) {
      v16h b = load_b_h(w2t, KD2, nBase + j * 16, k0, lane);
      acc[0][j] = __builtin_amdgcn_wmma_f32_16x16x32_f16(
          false, a0, false, b, (short)0, acc[0][j], false, false);
      acc[1][j] = __builtin_amdgcn_wmma_f32_16x16x32_f16(
          false, a1, false, b, (short)0, acc[1][j], false, false);
    }
  }

  const int hf = lane >> 4;
  const int nl = lane & 15;
  float biasv[4], w3v[4];
#pragma unroll
  for (int j = 0; j < 4; ++j) {
    const int col = nBase + j * 16 + nl;
    biasv[j] = b2[col];
    w3v[j]   = w3[col];
  }
#pragma unroll
  for (int mi = 0; mi < 2; ++mi)
#pragma unroll
    for (int r = 0; r < 8; ++r) {
      float s = 0.f;
#pragma unroll
      for (int j = 0; j < 4; ++j) {
        float v = acc[mi][j][r] + biasv[j];
        v = v > 0.f ? v : 0.f;           // relu
        s += v * w3v[j];                 // partial dot with W3
      }
      // reduce across the 16 lanes holding the same row (masks 1,2,4,8 stay
      // inside each 16-lane half of the wave32)
      s += __shfl_xor(s, 1, 32);
      s += __shfl_xor(s, 2, 32);
      s += __shfl_xor(s, 4, 32);
      s += __shfl_xor(s, 8, 32);
      if (nl == 0) {
        const int row = mBase + mi * 16 + hf * 8 + r;
        atomicAdd(&logits[row], s);
      }
    }
}

// ---------------- softmax over K=8 + mask ----------------------------------
__global__ __launch_bounds__(256)
void softmax_kernel(const float* __restrict__ logits,
                    const int* __restrict__ mask,
                    const float* __restrict__ b3, float* __restrict__ out) {
  const int g = blockIdx.x * blockDim.x + threadIdx.x;   // over BB*SS
  if (g >= BB * SS) return;
  const int base = g * KC;
  const float bias = b3[0];
  float l[KC];
  float m = -3.4e38f;
#pragma unroll
  for (int j = 0; j < KC; ++j) {
    l[j] = logits[base + j] + bias;
    m = fmaxf(m, l[j]);
  }
  float sum = 0.f;
#pragma unroll
  for (int j = 0; j < KC; ++j) { l[j] = expf(l[j] - m); sum += l[j]; }
  const float inv = 1.f / sum;
#pragma unroll
  for (int j = 0; j < KC; ++j)
    out[base + j] = l[j] * inv * (float)mask[base + j];
}

// ---------------------------------------------------------------------------
extern "C" void kernel_launch(void* const* d_in, const int* in_sizes, int n_in,
                              void* d_out, int out_size, void* d_ws, size_t ws_size,
                              hipStream_t stream) {
  (void)in_sizes; (void)n_in; (void)out_size; (void)ws_size;

  const float* x    = (const float*)d_in[0];
  const int*   mask = (const int*)  d_in[1];
  const float* W1   = (const float*)d_in[2];
  const float* b1   = (const float*)d_in[3];
  const float* W2   = (const float*)d_in[4];
  const float* b2   = (const float*)d_in[5];
  const float* W3   = (const float*)d_in[6];
  const float* b3   = (const float*)d_in[7];
  float* out        = (float*)d_out;

  // workspace layout (256B aligned)
  char* ws = (char*)d_ws;
  size_t off = 0;
  auto alloc = [&](size_t bytes) {
    char* p = ws + off;
    off += (bytes + 255) & ~(size_t)255;
    return p;
  };
  _Float16* w1t    = (_Float16*)alloc((size_t)HDIM * KD1 * sizeof(_Float16));
  _Float16* w2t    = (_Float16*)alloc((size_t)HDIM * KD2 * sizeof(_Float16));
  float*    sent   = (float*)   alloc((size_t)MROWS * HDIM * sizeof(float));
  _Float16* cur    = (_Float16*)alloc((size_t)MROWS * KD2 * sizeof(_Float16));
  float*    logits = (float*)   alloc((size_t)MROWS * sizeof(float));

  // 1. weight prep: transpose + convert to f16 (weights then live in L2)
  transpose_to_half_kernel<<<dim3(HDIM / 32, KD1 / 32), dim3(32, 8), 0, stream>>>(
      W1, w1t, KD1, HDIM);
  transpose_to_half_kernel<<<dim3(HDIM / 32, KD2 / 32), dim3(32, 8), 0, stream>>>(
      W2, w2t, KD2, HDIM);
  hipMemsetAsync(logits, 0, (size_t)MROWS * sizeof(float), stream);

  // 2. sent = x @ W1 + b1
  gemm1_kernel<<<dim3(HDIM / 128, MROWS / 128), 256, 0, stream>>>(
      x, w1t, b1, sent);

  // 3. exclusive cummax along S -> cur = [f16(sent) | f16(before)]
  cummax_kernel<<<(BB * KC * HDIM) / 256, 256, 0, stream>>>(sent, cur);

  // 4. logits += relu(cur @ W2 + b2) . W3
  gemm2_kernel<<<dim3(HDIM / 128, MROWS / 128), 256, 0, stream>>>(
      cur, w2t, b2, W3, logits);

  // 5. softmax over K classes, apply mask
  softmax_kernel<<<(BB * SS + 255) / 256, 256, 0, stream>>>(
      logits, mask, b3, out);
}
